// SetEncoder_9113920602243
// MI455X (gfx1250) — matrix operand order repaired and checked by reading, not compiled, not run on hardware
//
#include <hip/hip_runtime.h>

// ---------------------------------------------------------------------------
// MI455X (gfx1250) Set-Transformer encoder forward.
// bf16 WMMA (v_wmma_f32_16x16x32_bf16) for all GEMMs + fused flash-style
// attention (no materialized score tensor), wave32 throughout.
// Weights pre-transposed+padded at conversion: GEMM staging is pure b128
// (global and LDS), no scalar stores in the hot loop.
// ---------------------------------------------------------------------------

typedef __attribute__((ext_vector_type(16))) __bf16 v16bf;
typedef __attribute__((ext_vector_type(8)))  float  v8f;
typedef __attribute__((ext_vector_type(4)))  unsigned v4u;

union Frag16 {
  v16bf v;
  unsigned u[8];
  unsigned short s[16];
  v4u q[2];
};
union H8 { unsigned short s[8]; v4u q; };

static __device__ __forceinline__ unsigned short f2bfu(float f) {
  union { float f; unsigned u; } x; x.f = f;
  unsigned r = x.u + 0x7FFFu + ((x.u >> 16) & 1u);  // round-to-nearest-even
  return (unsigned short)(r >> 16);
}
static __device__ __forceinline__ float bfu2f(unsigned short h) {
  union { unsigned u; float f; } x; x.u = ((unsigned)h) << 16;
  return x.f;
}
static __device__ __forceinline__ v8f wmma_bf16(const Frag16& a, const Frag16& b, v8f c) {
  return __builtin_amdgcn_wmma_f32_16x16x32_bf16(false, a.v, false, b.v,
                                                 (short)0, c, false, false);
}

// ---------------------------------------------------------------------------
// fp32 -> bf16 plain conversion (inducing points / seeds: GEMM A operands)
// ---------------------------------------------------------------------------
__global__ void cvt_bf16_kernel(const float* __restrict__ src,
                                unsigned short* __restrict__ dst, int ntot) {
  int i = blockIdx.x * blockDim.x + threadIdx.x;
  if (i < ntot) dst[i] = f2bfu(src[i]);
}

// ---------------------------------------------------------------------------
// fp32 [K][512] -> bf16 W^T [512][Kpad], zero-padded k tail.  One-off pass so
// every GEMM B-fragment is k-contiguous (b128 staging, no LDS transpose).
// ---------------------------------------------------------------------------
__global__ void cvtT_bf16_kernel(const float* __restrict__ src,
                                 unsigned short* __restrict__ dst,
                                 int Ksrc, int Kpad, int ntot) {
  int i = blockIdx.x * blockDim.x + threadIdx.x;
  if (i >= ntot) return;                    // ntot = 512 * Kpad
  int n = i / Kpad, k = i - n * Kpad;
  dst[i] = (k < Ksrc) ? f2bfu(src[(long)k * 512 + n]) : (unsigned short)0;
}

// ---------------------------------------------------------------------------
// float2bit featurizer: x[B,N,3] fp32 -> h[B,N,64] bf16, cols 0..47 in
// {+1,-1}, cols 48..63 zero-padded (so GEMM K=64, pure vector loads).
// ---------------------------------------------------------------------------
__global__ void bits_kernel(const float* __restrict__ x,
                            unsigned short* __restrict__ out, int total) {
  int idx = blockIdx.x * blockDim.x + threadIdx.x;
  if (idx >= total) return;
  int col = idx & 63;
  if (col >= 48) { out[idx] = 0; return; }
  int row = idx >> 6;                 // b*N + n
  int c = col >> 4, bit = col & 15;
  float f = x[row * 3 + c];
  float bv;
  if (bit == 0) {
    bv = (f + 0.001f) < 0.f ? 1.f : 0.f;       // sign bit
  } else {
    float f1   = fabsf(f);
    float esci = (f1 == 0.f) ? -15.f : floorf(log2f(f1));
    if (bit <= 5) {                             // 5 exponent bits, MSB first
      float edec = esci + 15.f;
      float div  = exp2f((float)(5 - bit));
      bv = fmodf(floorf(edec / div), 2.f);
    } else {                                    // 10 mantissa bits
      int j = bit - 6;
      float f2  = (f1 == 0.f) ? 0.f : f1 * exp2f(-esci);
      float rem = f2 - floorf(f2);
      float t   = rem * exp2f((float)j);
      t = t - floorf(t);
      bv = floorf(2.f * t);
    }
  }
  out[idx] = (bv > 0.5f) ? (unsigned short)0x3F80u : (unsigned short)0xBF80u;
}

// ---------------------------------------------------------------------------
// Tiled bf16 WMMA GEMM:  C[b] = act( A[b] @ W + bias ),  K % 32 == 0,
// W supplied TRANSPOSED as Wt[512][K] (k-contiguous).
//   mode 0: C = A@W + bias                  (row-major [M][512])
//   mode 1: C = R + relu(A@W + bias)        (row-major, MLP residual)
//   mode 2: C^T = (A@W + bias)^T            ([512][M], packed b128 stores;
//                                            used for V so attention reads
//                                            contiguous B-fragments)
// WG = 256 threads / 8 waves; tile 64(M) x 64(N); 2 WMMA per wave per chunk.
// ---------------------------------------------------------------------------
__global__ __launch_bounds__(256) void gemm_bf16_kernel(
    const unsigned short* __restrict__ A, long aBatch, int lda,
    const unsigned short* __restrict__ Wt,           // [512][K]
    const float* __restrict__ bias,
    const unsigned short* __restrict__ R, long rBatch, int ldr,
    unsigned short* __restrict__ C, long cBatch, int ldc,
    int M, int K, int mode) {
  __shared__ __align__(16) unsigned short As[64 * 32];   // [m][k]
  __shared__ __align__(16) unsigned short WsT[64 * 32];  // [n][k]

  const int b = blockIdx.z;
  const unsigned short* Ab = A + (long)b * aBatch;
  const unsigned short* Rb = (mode == 1) ? (R + (long)b * rBatch) : nullptr;
  unsigned short* Cb = C + (long)b * cBatch;
  const int mBase = blockIdx.y * 64;
  const int nBase = blockIdx.x * 64;
  const int t = threadIdx.x;
  const int lane = t & 31;
  const int wave = t >> 5;
  const int waveM = wave >> 1, waveN = wave & 1;  // 4 x 2 waves, 2 n-tiles each
  const int hv = lane >> 4, l16 = lane & 15;

  // staging addresses: both A and Wt tiles are 64 rows x 32 k, one b128 each
  const int srow = t >> 2, scol = (t & 3) << 3;
  int agr = mBase + srow; if (agr >= M) agr = M - 1;     // clamp (M=16 tiles)
  const unsigned short* aSrc = Ab + (long)agr * lda + scol;
  const unsigned short* wSrc = Wt + (long)(nBase + srow) * K + scol;

  v8f acc0 = {0.f, 0.f, 0.f, 0.f, 0.f, 0.f, 0.f, 0.f};
  v8f acc1 = acc0;

  for (int k0 = 0; k0 < K; k0 += 32) {
    *(v4u*)&As[srow * 32 + scol]  = *(const v4u*)(aSrc + k0);
    *(v4u*)&WsT[srow * 32 + scol] = *(const v4u*)(wSrc + k0);
    __builtin_prefetch(aSrc + k0 + 32, 0, 1);   // global_prefetch_b8
    __builtin_prefetch(wSrc + k0 + 32, 0, 1);
    __syncthreads();

    Frag16 af, bf0, bf1;
    const unsigned* As32 = (const unsigned*)As;          // row = 16 dwords
    const unsigned* Ws32 = (const unsigned*)WsT;
    const int am = waveM * 16 + l16;
    const int bn = waveN * 32 + l16;
#pragma unroll
    for (int v = 0; v < 8; ++v) {
      int dw = (v < 4 ? v : v + 4) + hv * 4;             // A-frag K pattern
      af.u[v]  = As32[am * 16 + dw];
      bf0.u[v] = Ws32[bn * 16 + dw];                     // B-frag: same K pattern as A
      bf1.u[v] = Ws32[(bn + 16) * 16 + dw];
    }
    acc0 = wmma_bf16(af, bf0, acc0);
    acc1 = wmma_bf16(af, bf1, acc1);
    __syncthreads();
  }

  const int gn0 = nBase + waveN * 32 + l16;
  const int gn1 = gn0 + 16;
  const float bv0 = bias[gn0], bv1 = bias[gn1];
  if (mode == 2) {            // transposed store: 8 contiguous rows -> b128
    const int gm = mBase + waveM * 16 + hv * 8;
    H8 h0, h1;
#pragma unroll
    for (int r = 0; r < 8; ++r) {
      h0.s[r] = f2bfu(acc0[r] + bv0);
      h1.s[r] = f2bfu(acc1[r] + bv1);
    }
    *(v4u*)(Cb + (long)gn0 * ldc + gm) = h0.q;
    *(v4u*)(Cb + (long)gn1 * ldc + gm) = h1.q;
  } else {
#pragma unroll
    for (int r = 0; r < 8; ++r) {
      int gm = mBase + waveM * 16 + hv * 8 + r;          // C row = r + 8*half
      if (gm < M) {
        float v0 = acc0[r] + bv0, v1 = acc1[r] + bv1;
        if (mode == 1) {
          v0 = bfu2f(Rb[(long)gm * ldr + gn0]) + (v0 > 0.f ? v0 : 0.f);
          v1 = bfu2f(Rb[(long)gm * ldr + gn1]) + (v1 > 0.f ? v1 : 0.f);
        }
        Cb[(long)gm * ldc + gn0] = f2bfu(v0);
        Cb[(long)gm * ldc + gn1] = f2bfu(v1);
      }
    }
  }
}

// ---------------------------------------------------------------------------
// Fused multi-head attention (flash-style online softmax), one wave per
// (batch, head, 16-query tile).  D=512, H=8, Dh=64.  V is TRANSPOSED
// ([512][Nk] per batch) so B-fragments are contiguous v4u loads.
//   out = Q_ + softmax(Q_h K_h^T * scale) V_h           (residual fused)
// ---------------------------------------------------------------------------
__global__ __launch_bounds__(32) void attn_kernel(
    const unsigned short* __restrict__ Q, long qBatch,
    const unsigned short* __restrict__ Km, long kBatch,
    const unsigned short* __restrict__ Vt, long vBatch,
    unsigned short* __restrict__ O, long oBatch,
    int Nk, float scale) {
  __shared__ __align__(16) unsigned short Ps[16 * 32];   // P tile transpose

  const int b = blockIdx.z, h = blockIdx.y;
  const int qb = blockIdx.x * 16;
  const int lane = threadIdx.x & 31;
  const int hv = lane >> 4, l16 = lane & 15;
  const int cb = h * 64;
  const unsigned short* Qb = Q + (long)b * qBatch;
  const unsigned short* Kb = Km + (long)b * kBatch;
  const unsigned short* Vb = Vt + (long)b * vBatch;
  unsigned short* Ob = O + (long)b * oBatch;

  // Q fragments for k-dims [0,32) and [32,64) of this head
  Frag16 qa[2];
  {
    const unsigned* qrow = (const unsigned*)(Qb + (long)(qb + l16) * 512 + cb);
#pragma unroll
    for (int c = 0; c < 2; ++c)
#pragma unroll
      for (int v = 0; v < 8; ++v)
        qa[c].u[v] = qrow[c * 16 + (v < 4 ? v : v + 4) + hv * 4];
  }

  float rmax[8], rsum[8];
  v8f oacc[4];
  const v8f vzero = {0.f, 0.f, 0.f, 0.f, 0.f, 0.f, 0.f, 0.f};
#pragma unroll
  for (int r = 0; r < 8; ++r) { rmax[r] = -1e30f; rsum[r] = 0.f; }
#pragma unroll
  for (int n = 0; n < 4; ++n) oacc[n] = vzero;

  for (int kb = 0; kb < Nk; kb += 32) {
    if (kb + 32 < Nk)                         // warm L0 for next key block
      __builtin_prefetch(Kb + (long)(kb + 32 + l16) * 512 + cb, 0, 1);
    // ---- scores: two 16x16 sub-tiles over 32 keys -------------------------
    v8f s[2];
#pragma unroll
    for (int sub = 0; sub < 2; ++sub) {
      const v4u* krow =
          (const v4u*)(Kb + (long)(kb + sub * 16 + l16) * 512 + cb);
      v8f z = vzero;
#pragma unroll
      for (int c = 0; c < 2; ++c) {
        Frag16 kf;                        // B-frag: A-frag K pattern
        kf.q[0] = krow[c * 4 + hv];
        kf.q[1] = krow[c * 4 + 2 + hv];
        z = wmma_bf16(qa[c], kf, z);
      }
#pragma unroll
      for (int r = 0; r < 8; ++r) s[sub][r] = z[r] * scale;
    }
    // ---- online softmax row stats (row m = r + 8*hv, cols across lanes) ---
    float fac[8];
#pragma unroll
    for (int r = 0; r < 8; ++r) {
      float m2 = fmaxf(s[0][r], s[1][r]);
#pragma unroll
      for (int xm = 1; xm < 16; xm <<= 1) m2 = fmaxf(m2, __shfl_xor(m2, xm, 32));
      float nmax = fmaxf(rmax[r], m2);
      fac[r] = __expf(rmax[r] - nmax);
      rmax[r] = nmax;
      rsum[r] *= fac[r];
    }
#pragma unroll
    for (int n = 0; n < 4; ++n)
#pragma unroll
      for (int r = 0; r < 8; ++r) oacc[n][r] *= fac[r];

    // ---- P = exp(S - max): transpose through LDS into A-frag --------------
    asm volatile("s_wait_dscnt 0" ::: "memory");  // prior P reads drained
#pragma unroll
    for (int sub = 0; sub < 2; ++sub)
#pragma unroll
      for (int r = 0; r < 8; ++r) {
        float p = __expf(s[sub][r] - rmax[r]);
        float ps = p;
#pragma unroll
        for (int xm = 1; xm < 16; xm <<= 1) ps += __shfl_xor(ps, xm, 32);
        rsum[r] += ps;
        Ps[(hv * 8 + r) * 32 + sub * 16 + l16] = f2bfu(p);
      }
    asm volatile("s_wait_dscnt 0" ::: "memory");  // LDS in-order per wave

    Frag16 pf;
    const unsigned* Ps32 = (const unsigned*)Ps;
#pragma unroll
    for (int v = 0; v < 8; ++v)
      pf.u[v] = Ps32[l16 * 16 + hv * 8 + v];

    // ---- O += P @ V  (V^T: contiguous keys per lane -> 2 x b128) ----------
#pragma unroll
    for (int n = 0; n < 4; ++n) {
      const int dh = cb + n * 16 + l16;
      const v4u* vrow = (const v4u*)(Vb + (long)dh * Nk + kb + hv * 16);
      Frag16 vf;
      vf.q[0] = vrow[0];
      vf.q[1] = vrow[1];
      oacc[n] = wmma_bf16(pf, vf, oacc[n]);
    }
  }

  // ---- epilogue: out = Q_ + O / rowsum ------------------------------------
#pragma unroll
  for (int n = 0; n < 4; ++n) {
    int dh = cb + n * 16 + l16;
#pragma unroll
    for (int r = 0; r < 8; ++r) {
      int m = qb + hv * 8 + r;
      float qv = bfu2f(Qb[(long)m * 512 + dh]);
      float val = qv + oacc[n][r] / rsum[r];
      Ob[(long)m * 512 + dh] = f2bfu(val);
    }
  }
}

// ---------------------------------------------------------------------------
// LayerNorm over D=512, one wave per row, 16 contiguous channels per lane
// (b128 loads).  f32out=1 writes fp32 (final output).
// ---------------------------------------------------------------------------
__global__ __launch_bounds__(256) void ln_kernel(
    const unsigned short* __restrict__ X,
    const float* __restrict__ g, const float* __restrict__ bta,
    void* __restrict__ out, int rows, int f32out) {
  int wid = threadIdx.x >> 5, lane = threadIdx.x & 31;
  long row = (long)blockIdx.x * 8 + wid;
  if (row >= rows) return;
  const unsigned short* xr = X + row * 512 + lane * 16;
  H8 a, c;
  a.q = *(const v4u*)xr;
  c.q = *(const v4u*)(xr + 8);
  float v[16];
  float s = 0.f;
#pragma unroll
  for (int i = 0; i < 8; ++i) { v[i] = bfu2f(a.s[i]); v[i + 8] = bfu2f(c.s[i]); }
#pragma unroll
  for (int i = 0; i < 16; ++i) s += v[i];
#pragma unroll
  for (int xm = 1; xm < 32; xm <<= 1) s += __shfl_xor(s, xm, 32);
  float mu = s * (1.f / 512.f);
  float q = 0.f;
#pragma unroll
  for (int i = 0; i < 16; ++i) { float d = v[i] - mu; q += d * d; }
#pragma unroll
  for (int xm = 1; xm < 32; xm <<= 1) q += __shfl_xor(q, xm, 32);
  float rstd = rsqrtf(q * (1.f / 512.f) + 1e-5f);
  const int c0 = lane * 16;
  if (f32out) {
    float* o = (float*)out + row * 512 + c0;
#pragma unroll
    for (int i = 0; i < 16; ++i) o[i] = (v[i] - mu) * rstd * g[c0 + i] + bta[c0 + i];
  } else {
    H8 h0, h1;
#pragma unroll
    for (int i = 0; i < 8; ++i) {
      h0.s[i] = f2bfu((v[i] - mu) * rstd * g[c0 + i] + bta[c0 + i]);
      h1.s[i] = f2bfu((v[i + 8] - mu) * rstd * g[c0 + 8 + i] + bta[c0 + 8 + i]);
    }
    unsigned short* o = (unsigned short*)out + row * 512 + c0;
    *(v4u*)o = h0.q;
    *(v4u*)(o + 8) = h1.q;
  }
}

// ---------------------------------------------------------------------------
// Host orchestration
// ---------------------------------------------------------------------------
extern "C" void kernel_launch(void* const* d_in, const int* in_sizes, int n_in,
                              void* d_out, int out_size, void* d_ws, size_t ws_size,
                              hipStream_t stream) {
  (void)in_sizes; (void)n_in; (void)out_size; (void)ws_size;
  typedef unsigned short u16;
  const int B = 32, N = 1024, D = 512, MI = 64, SE = 16;
  const float SCALE = 0.0441941738241592f;  // 1/sqrt(512)

  size_t off = 0;
  auto alloc = [&](size_t bytes) -> void* {
    off = (off + 255) & ~(size_t)255;
    void* p = (char*)d_ws + off;
    off += bytes;
    return p;
  };
  u16* BA = (u16*)alloc((size_t)B * N * D * 2);   // rotating big buffers
  u16* BB = (u16*)alloc((size_t)B * N * D * 2);
  u16* BC = (u16*)alloc((size_t)B * N * D * 2);
  u16* s0 = (u16*)alloc((size_t)B * MI * D * 2);  // inducing-point pool
  u16* s1 = (u16*)alloc((size_t)B * MI * D * 2);
  u16* s2 = (u16*)alloc((size_t)B * MI * D * 2);
  u16* s3 = (u16*)alloc((size_t)B * MI * D * 2);
  u16* sQ = (u16*)alloc((size_t)MI * D * 2);      // batch-shared Q_ (I / S)

  // plain convert (A-side operands: inducing points, seeds)
  auto cvtA = [&](int leaf, size_t ntot) -> u16* {
    u16* dst = (u16*)alloc(ntot * 2);
    cvt_bf16_kernel<<<(int)((ntot + 255) / 256), 256, 0, stream>>>(
        (const float*)d_in[leaf], dst, (int)ntot);
    return dst;
  };
  // transpose+pad convert (weights: [Ksrc][512] -> bf16 [512][Kpad])
  auto cvtW = [&](int leaf, int Ksrc, int Kpad) -> u16* {
    size_t ntot = (size_t)512 * Kpad;
    u16* dst = (u16*)alloc(ntot * 2);
    cvtT_bf16_kernel<<<(int)((ntot + 255) / 256), 256, 0, stream>>>(
        (const float*)d_in[leaf], dst, Ksrc, Kpad, (int)ntot);
    return dst;
  };

  struct MabW {
    u16 *qW, *kW, *vW, *oW;
    const float *qb, *kb, *vb, *ob, *g0, *b0, *g1, *b1;
  };
  // leaf order inside a _mab: qW,qb,kW,kb,vW,vb,oW,ob,g0,b0,g1,b1
  auto loadMab = [&](int base, int dqS, int dqP, int dkS, int dkP) -> MabW {
    MabW m;
    m.qW = cvtW(base + 0, dqS, dqP);  m.qb = (const float*)d_in[base + 1];
    m.kW = cvtW(base + 2, dkS, dkP);  m.kb = (const float*)d_in[base + 3];
    m.vW = cvtW(base + 4, dkS, dkP);  m.vb = (const float*)d_in[base + 5];
    m.oW = cvtW(base + 6, D, D);      m.ob = (const float*)d_in[base + 7];
    m.g0 = (const float*)d_in[base + 8];  m.b0 = (const float*)d_in[base + 9];
    m.g1 = (const float*)d_in[base + 10]; m.b1 = (const float*)d_in[base + 11];
    return m;
  };

  auto gemm = [&](const u16* A, long aB, int lda, const u16* Wt,
                  const float* bias, const u16* R, long rB,
                  u16* C, long cB, int ldc, int M, int K, int batches, int mode) {
    dim3 g(D / 64, (M + 63) / 64, batches);
    gemm_bf16_kernel<<<g, 256, 0, stream>>>(A, aB, lda, Wt, bias,
                                            R, rB, D, C, cB, ldc, M, K, mode);
  };
  auto attn = [&](const u16* Q, long qB, const u16* K_, long kB,
                  const u16* V_, long vB, u16* O, long oB, int Nq, int Nk) {
    dim3 g(Nq / 16, 8, B);
    attn_kernel<<<g, 32, 0, stream>>>(Q, qB, K_, kB, V_, vB, O, oB, Nk, SCALE);
  };
  auto ln = [&](const u16* X, const float* g_, const float* b_,
                void* out, int rows, int f32o) {
    ln_kernel<<<rows / 8, 256, 0, stream>>>(X, g_, b_, out, rows, f32o);
  };

  // isab leaf layout: I at base, mab0 at base+1, mab1 at base+13 (25 leaves)
  auto runIsab = [&](int base, int dinS, int dinP) {
    u16* Ibf = cvtA(base, (size_t)MI * D);
    MabW m0 = loadMab(base + 1, D, D, dinS, dinP);   // MAB(I, X)
    MabW m1 = loadMab(base + 13, dinS, dinP, D, D);  // MAB(X, H)
    long xs = (long)N * dinP, bs = (long)N * D, ss = (long)MI * D;
    // ---- mab0: H = MAB(I, X) ----
    gemm(Ibf, 0, D, m0.qW, m0.qb, nullptr, 0, sQ, 0, D, MI, D, 1, 0);
    gemm(BA, xs, dinP, m0.kW, m0.kb, nullptr, 0, BB, bs, D, N, dinP, B, 0);
    gemm(BA, xs, dinP, m0.vW, m0.vb, nullptr, 0, BC, bs, N, N, dinP, B, 2); // V^T [512][N]
    attn(sQ, 0, BB, bs, BC, bs, s0, ss, MI, N);
    ln(s0, m0.g0, m0.b0, s0, B * MI, 0);
    gemm(s0, ss, D, m0.oW, m0.ob, s0, ss, s1, ss, D, MI, D, B, 1);
    ln(s1, m0.g1, m0.b1, s1, B * MI, 0);                   // Hm = s1
    // ---- mab1: X' = MAB(X, Hm) ----
    gemm(BA, xs, dinP, m1.qW, m1.qb, nullptr, 0, BB, bs, D, N, dinP, B, 0);
    gemm(s1, ss, D, m1.kW, m1.kb, nullptr, 0, s2, ss, D, MI, D, B, 0);
    gemm(s1, ss, D, m1.vW, m1.vb, nullptr, 0, s3, ss, MI, MI, D, B, 2);     // V^T [512][64]
    attn(BB, bs, s2, ss, s3, ss, BC, bs, N, MI);
    ln(BC, m1.g0, m1.b0, BC, B * N, 0);
    gemm(BC, bs, D, m1.oW, m1.ob, BC, bs, BA, bs, D, N, D, B, 1);
    ln(BA, m1.g1, m1.b1, BA, B * N, 0);                    // new X in BA
  };

  // ---- float2bit featurizer: X0 in BA as [B,N,64] bf16 (48 + 16 zero) ----
  bits_kernel<<<(B * N * 64) / 256, 256, 0, stream>>>(
      (const float*)d_in[0], BA, B * N * 64);

  // ---- encoder stack (leaf bases per insertion-order flattening) ----
  runIsab(1, 48, 64);    // isab0: 48 (padded to 64) -> 512
  runIsab(26, D, D);
  runIsab(51, D, D);
  runIsab(76, D, D);

  // ---- PMA: out = MAB(S, X) -> [B,16,512] fp32 ----
  u16* Sbf = cvtA(101, (size_t)SE * D);
  MabW pm = loadMab(102, D, D, D, D);
  long bs = (long)N * D, ps = (long)SE * D;
  gemm(Sbf, 0, D, pm.qW, pm.qb, nullptr, 0, sQ, 0, D, SE, D, 1, 0);
  gemm(BA, bs, D, pm.kW, pm.kb, nullptr, 0, BB, bs, D, N, D, B, 0);
  gemm(BA, bs, D, pm.vW, pm.vb, nullptr, 0, BC, bs, N, N, D, B, 2);         // V^T
  attn(sQ, 0, BB, bs, BC, bs, s0, ps, SE, N);
  ln(s0, pm.g0, pm.b0, s0, B * SE, 0);
  gemm(s0, ps, D, pm.oW, pm.ob, s0, ps, s1, ps, D, SE, D, B, 1);
  ln(s1, pm.g1, pm.b1, d_out, B * SE, 1);   // fp32 out -> d_out
}
